// RRN_12841952215130
// MI455X (gfx1250) — matrix operand order repaired and checked by reading, not compiled, not run on hardware
//
#include <hip/hip_runtime.h>
#include <math.h>

typedef __attribute__((ext_vector_type(16))) _Float16 v16h;
typedef __attribute__((ext_vector_type(8)))  _Float16 v8h;
typedef __attribute__((ext_vector_type(8)))  float    v8f;

#define GDIM 8
#define NCELL 64
#define DEG 17
#define STEPS 32
#define HDIM 96
#define BSZ 16
#define BN (BSZ*NCELL)          /* 1024 */
#define ROWS_MSG (BN*DEG)       /* 17408 */

__device__ __forceinline__ v8f wmma_f16(v16h a, v16h b, v8f c) {
    return __builtin_amdgcn_wmma_f32_16x16x32_f16(false, a, false, b, (short)0, c,
                                                  false, false);
}

// A-fragment (16-bit A 16x32): lane holds runs [off, off+8) and [off+16, off+24)
// of its row, off = k0 + ((lane&16)?8:0).  Both runs are 16B aligned.
__device__ __forceinline__ v16h load_afrag(const _Float16* __restrict__ p, int off) {
    v8h lo = *(const v8h*)(p + off);
    v8h hi = *(const v8h*)(p + off + 16);
    return __builtin_shufflevector(lo, hi, 0,1,2,3,4,5,6,7,8,9,10,11,12,13,14,15);
}

__device__ __forceinline__ v16h load_bfrag(const _Float16* __restrict__ Wp,
                                           int blk, int lane) {
    return *(const v16h*)(Wp + (((size_t)blk) * 32 + lane) * 16);
}

__device__ __forceinline__ float sigmf(float x) { return 1.f / (1.f + expf(-x)); }

// ---------------------------------------------------------------------------
// Pack a row-major f32 weight [kreal, nreal] (ld = ldw) into f16 WMMA
// B-fragment layout, zero-padded to Kchunks*32 x Ntiles*16.
// ---------------------------------------------------------------------------
__global__ void pack_w_kernel(const float* __restrict__ W, int ldw, int kreal,
                              int nreal, int Kchunks, int Ntiles,
                              _Float16* __restrict__ out)
{
    int idx = blockIdx.x * blockDim.x + threadIdx.x;
    int total = Kchunks * Ntiles * 512;
    if (idx >= total) return;
    int i    = idx & 15;
    int lane = (idx >> 4) & 31;
    int blk  = idx >> 9;
    int nt   = blk % Ntiles;
    int kt   = blk / Ntiles;
    int K    = kt * 32 + ((lane & 16) ? 8 : 0) + (i & 7) + ((i & 8) ? 16 : 0);
    int n    = nt * 16 + (lane & 15);
    float v  = (K < kreal && n < nreal) ? W[(size_t)K * ldw + n] : 0.f;
    out[idx] = (_Float16)v;
}

__global__ void pad_bias_kernel(const float* __restrict__ b, int nreal,
                                float* __restrict__ out)
{
    int i = threadIdx.x;
    out[i] = (i < nreal) ? b[i] : 0.f;
}

// ---------------------------------------------------------------------------
// Generic GEMM: Y = act(X @ Wp + bias*scale);  X f16 row-major, Wp packed f16.
// Output to f16 (Yh) when feeding another WMMA stage, else f32 (Yf).
// ---------------------------------------------------------------------------
__global__ __launch_bounds__(256) void gemm_kernel(
    const _Float16* __restrict__ X, int ldx,
    const _Float16* __restrict__ Wp,
    const float* __restrict__ bias, float bscale,
    float* __restrict__ Yf, _Float16* __restrict__ Yh, int ldy,
    int Mtiles, int Ntiles, int Kchunks, int do_relu)
{
    int wid = blockIdx.x * 8 + (threadIdx.x >> 5);
    if (wid >= Mtiles * Ntiles) return;
    int lane  = threadIdx.x & 31;
    int mt    = wid / Ntiles, nt = wid % Ntiles;
    int r0    = mt * 16;
    const _Float16* xr = X + (size_t)(r0 + (lane & 15)) * ldx;
    int abase = (lane & 16) ? 8 : 0;

    v8f c = {0.f,0.f,0.f,0.f,0.f,0.f,0.f,0.f};
    for (int kc = 0; kc < Kchunks; ++kc) {
        v16h a = load_afrag(xr, kc * 32 + abase);
        c = wmma_f16(a, load_bfrag(Wp, kc * Ntiles + nt, lane), c);
    }
    float bv  = bias[nt * 16 + (lane & 15)] * bscale;
    int   mhi = (lane & 16) ? 8 : 0;
    int   col = nt * 16 + (lane & 15);
#pragma unroll
    for (int rI = 0; rI < 8; ++rI) {
        float v = c[rI] + bv;
        if (do_relu) v = fmaxf(v, 0.f);
        size_t off = (size_t)(r0 + rI + mhi) * ldy + col;
        if (Yh) Yh[off] = (_Float16)v;
        else    Yf[off] = v;
    }
}

// ---------------------------------------------------------------------------
// Fused message MLP (layers 0..2 with relu): rows = (b, j, d); row content is
// concat(h16[b, edges[j,d]], h16[b, j]).  Outputs f16 t2 rows.
// ---------------------------------------------------------------------------
__global__ __launch_bounds__(256) void msg_mlp_kernel(
    const _Float16* __restrict__ h16, const int* __restrict__ edges,
    const _Float16* __restrict__ Wp0, const float* __restrict__ msg0_b,
    const _Float16* __restrict__ Wp1, const _Float16* __restrict__ Wp2,
    const float* __restrict__ msg_bs,
    _Float16* __restrict__ t2_rows)
{
    __shared__ __align__(16) _Float16 act[8][16][HDIM];  // per-wave 16x96 staging
    int wv   = threadIdx.x >> 5;
    int lane = threadIdx.x & 31;
    int wid  = blockIdx.x * 8 + wv;
    int r0   = wid * 16;
    int mrow = lane & 15;
    int r    = r0 + mrow;
    int bn   = r / DEG;
    int d    = r - bn * DEG;
    int j    = bn & (NCELL - 1);
    int b    = bn >> 6;
    int e    = edges[j * DEG + d];
    const _Float16* pe = h16 + (size_t)(b * NCELL + e) * HDIM;
    const _Float16* ps = h16 + (size_t)bn * HDIM;
    int abase = (lane & 16) ? 8 : 0;
    int ncol  = lane & 15;
    int mhi   = (lane & 16) ? 8 : 0;
    const v8f vz = {0.f,0.f,0.f,0.f,0.f,0.f,0.f,0.f};

    // ---- layer 0: [16x192] @ [192x96]; chunks 0-2 from edge row, 3-5 self ----
    v8f c[6];
#pragma unroll
    for (int nt = 0; nt < 6; ++nt) c[nt] = vz;
#pragma unroll
    for (int kc = 0; kc < 6; ++kc) {
        const _Float16* src = (kc < 3) ? pe : ps;
        int koff = ((kc < 3) ? kc : (kc - 3)) * 32 + abase;
        v16h a = load_afrag(src, koff);
#pragma unroll
        for (int nt = 0; nt < 6; ++nt)
            c[nt] = wmma_f16(a, load_bfrag(Wp0, kc * 6 + nt, lane), c[nt]);
    }
#pragma unroll
    for (int nt = 0; nt < 6; ++nt) {
        float bv = msg0_b[nt * 16 + ncol];
#pragma unroll
        for (int rI = 0; rI < 8; ++rI) {
            float v = fmaxf(c[nt][rI] + bv, 0.f);
            act[wv][rI + mhi][nt * 16 + ncol] = (_Float16)v;
        }
    }
    __syncthreads();

    // ---- layers 1 and 2: [16x96] @ [96x96] ----
    for (int layer = 0; layer < 2; ++layer) {
        const _Float16* Wp = (layer == 0) ? Wp1 : Wp2;
        const float*    bb = msg_bs + layer * HDIM;
        const _Float16* arow = &act[wv][mrow][0];
        v8f c2[6];
#pragma unroll
        for (int nt = 0; nt < 6; ++nt) c2[nt] = vz;
#pragma unroll
        for (int kc = 0; kc < 3; ++kc) {
            v16h a = load_afrag(arow, kc * 32 + abase);
#pragma unroll
            for (int nt = 0; nt < 6; ++nt)
                c2[nt] = wmma_f16(a, load_bfrag(Wp, kc * 6 + nt, lane), c2[nt]);
        }
        __syncthreads();
        if (layer == 0) {
#pragma unroll
            for (int nt = 0; nt < 6; ++nt) {
                float bv = bb[nt * 16 + ncol];
#pragma unroll
                for (int rI = 0; rI < 8; ++rI) {
                    float v = fmaxf(c2[nt][rI] + bv, 0.f);
                    act[wv][rI + mhi][nt * 16 + ncol] = (_Float16)v;
                }
            }
            __syncthreads();
        } else {
#pragma unroll
            for (int nt = 0; nt < 6; ++nt) {
                float bv = bb[nt * 16 + ncol];
#pragma unroll
                for (int rI = 0; rI < 8; ++rI) {
                    float v = fmaxf(c2[nt][rI] + bv, 0.f);
                    t2_rows[(size_t)(r0 + rI + mhi) * HDIM + nt * 16 + ncol] =
                        (_Float16)v;
                }
            }
        }
    }
}

// ---------------------------------------------------------------------------
// LSTM gates: gates = concat(msg,xin)[1024,192] @ W_ih + h[1024,96] @ W_hh + b
// ---------------------------------------------------------------------------
__global__ __launch_bounds__(256) void lstm_gates_kernel(
    const _Float16* __restrict__ msg, const _Float16* __restrict__ xin,
    const _Float16* __restrict__ h16,
    const _Float16* __restrict__ Wp_ih, const _Float16* __restrict__ Wp_hh,
    const float* __restrict__ b_ih, const float* __restrict__ b_hh,
    float* __restrict__ gates)
{
    int wid = blockIdx.x * 8 + (threadIdx.x >> 5);
    if (wid >= 64 * 24) return;
    int lane = threadIdx.x & 31;
    int mt = wid / 24, nt = wid % 24;
    int r0 = mt * 16;
    int row = r0 + (lane & 15);
    const _Float16* pm = msg + (size_t)row * HDIM;
    const _Float16* px = xin + (size_t)row * HDIM;
    const _Float16* ph = h16 + (size_t)row * HDIM;
    int abase = (lane & 16) ? 8 : 0;
    int n     = nt * 16 + (lane & 15);

    v8f c = {0.f,0.f,0.f,0.f,0.f,0.f,0.f,0.f};
#pragma unroll
    for (int kc = 0; kc < 6; ++kc) {
        const _Float16* src = (kc < 3) ? pm : px;
        int koff = ((kc < 3) ? kc : (kc - 3)) * 32 + abase;
        v16h a = load_afrag(src, koff);
        c = wmma_f16(a, load_bfrag(Wp_ih, kc * 24 + nt, lane), c);
    }
#pragma unroll
    for (int kc = 0; kc < 3; ++kc) {
        v16h a = load_afrag(ph, kc * 32 + abase);
        c = wmma_f16(a, load_bfrag(Wp_hh, kc * 24 + nt, lane), c);
    }
    float bv = b_ih[n] + b_hh[n];
    int mhi = (lane & 16) ? 8 : 0;
#pragma unroll
    for (int rI = 0; rI < 8; ++rI)
        gates[(size_t)(r0 + rI + mhi) * 384 + n] = c[rI] + bv;
}

// ---------------------------------------------------------------------------
// Small elementwise / reduction kernels
// ---------------------------------------------------------------------------
__global__ void encode_kernel(const int* __restrict__ x,
                              const float* __restrict__ digit_emb,
                              const float* __restrict__ row_emb,
                              const float* __restrict__ col_emb,
                              _Float16* __restrict__ feat)
{
    int idx = blockIdx.x * blockDim.x + threadIdx.x;
    if (idx >= BN * 64) return;
    int bn = idx >> 6, cc = idx & 63;
    int j = bn & (NCELL - 1);
    float v;
    if (cc < 16)       v = digit_emb[x[bn] * 16 + cc];
    else if (cc < 32)  v = row_emb[(j >> 3) * 16 + (cc - 16)];
    else if (cc < 48)  v = col_emb[(j & 7) * 16 + (cc - 32)];
    else               v = 0.f;
    feat[idx] = (_Float16)v;
}

__global__ void init_hs_kernel(_Float16* h16, float* s, int n) {
    int i = blockIdx.x * blockDim.x + threadIdx.x;
    if (i < n) { h16[i] = (_Float16)0.f; s[i] = 0.f; }
}

__global__ void degsum_kernel(const _Float16* __restrict__ t2,
                              _Float16* __restrict__ msgp)
{
    int idx = blockIdx.x * blockDim.x + threadIdx.x;
    if (idx >= BN * HDIM) return;
    int bn = idx / HDIM, cc = idx % HDIM;
    const _Float16* p = t2 + (size_t)(bn * DEG) * HDIM + cc;
    float acc = 0.f;
#pragma unroll
    for (int d = 0; d < DEG; ++d) acc += (float)p[(size_t)d * HDIM];
    msgp[idx] = (_Float16)acc;
}

__global__ void lstm_update_kernel(const float* __restrict__ gates,
                                   float* __restrict__ s,
                                   _Float16* __restrict__ h16)
{
    int idx = blockIdx.x * blockDim.x + threadIdx.x;
    if (idx >= BN * HDIM) return;
    int bn = idx / HDIM, cc = idx % HDIM;
    const float* g = gates + (size_t)bn * 384;
    float ig = sigmf(g[cc]);
    float fg = sigmf(g[HDIM + cc]);
    float gg = tanhf(g[2 * HDIM + cc]);
    float og = sigmf(g[3 * HDIM + cc]);
    float s2 = fg * s[idx] + ig * gg;
    s[idx] = s2;
    h16[idx] = (_Float16)(og * tanhf(s2));
}

__global__ __launch_bounds__(1024) void stats_kernel(
    const float* __restrict__ logits, const int* __restrict__ target,
    float* __restrict__ d_out, float* __restrict__ lossacc, int step)
{
    __shared__ float sl[1024];
    __shared__ int   sm[1024];
    __shared__ float sb[BSZ];
    int bn = threadIdx.x;
    const float* lg = logits + bn * 16;
    float m = lg[0]; int am = 0;
#pragma unroll
    for (int k = 1; k < GDIM; ++k) { float v = lg[k]; if (v > m) { m = v; am = k; } }
    float se = 0.f;
#pragma unroll
    for (int k = 0; k < GDIM; ++k) se += expf(lg[k] - m);
    float lz = m + logf(se);
    int tgt = target[bn] - 1;
    sl[bn] = -(lg[tgt] - lz);
    sm[bn] = (am == tgt) ? 1 : 0;
    if (step == STEPS - 1) d_out[1 + STEPS + bn] = (float)am;   // final_pred
    __syncthreads();
    if (bn < BSZ) {
        int all = 1;
        for (int jj = 0; jj < NCELL; ++jj) all &= sm[bn * NCELL + jj];
        sb[bn] = (float)all;
    }
    __syncthreads();
    for (int off = 512; off > 0; off >>= 1) {
        if (bn < off) sl[bn] += sl[bn + off];
        __syncthreads();
    }
    if (bn == 0) {
        float accv = 0.f;
        for (int bb = 0; bb < BSZ; ++bb) accv += sb[bb];
        d_out[1 + step] = accv / (float)BSZ;
        float lstep = sl[0] / (float)BN;
        float tot = (step == 0 ? 0.f : *lossacc) + lstep;
        *lossacc = tot;
        if (step == STEPS - 1) d_out[0] = tot / (float)STEPS;
    }
}

// ---------------------------------------------------------------------------
extern "C" void kernel_launch(void* const* d_in, const int* in_sizes, int n_in,
                              void* d_out, int out_size, void* d_ws, size_t ws_size,
                              hipStream_t stream)
{
    const int*   x         = (const int*)  d_in[0];
    const int*   target    = (const int*)  d_in[1];
    const int*   edges     = (const int*)  d_in[2];
    const float* digit_emb = (const float*)d_in[3];
    const float* row_emb   = (const float*)d_in[4];
    const float* col_emb   = (const float*)d_in[5];
    const float* in0_W     = (const float*)d_in[6];
    const float* in0_b     = (const float*)d_in[7];
    const float* in_Ws     = (const float*)d_in[8];
    const float* in_bs     = (const float*)d_in[9];
    const float* msg0_W    = (const float*)d_in[10];
    const float* msg0_b    = (const float*)d_in[11];
    const float* msg_Ws    = (const float*)d_in[12];
    const float* msg_bs    = (const float*)d_in[13];
    const float* W_ih      = (const float*)d_in[14];
    const float* W_hh      = (const float*)d_in[15];
    const float* b_ih      = (const float*)d_in[16];
    const float* b_hh      = (const float*)d_in[17];
    const float* pred_W    = (const float*)d_in[18];
    const float* pred_b    = (const float*)d_in[19];
    float* out = (float*)d_out;

    // f32 workspace region (all sizes multiples of 8 floats -> 32B alignment)
    float* ws    = (float*)d_ws;
    float* sbuf  = ws;  ws += BN * HDIM;
    float* gates = ws;  ws += BN * 4 * HDIM;
    float* logit = ws;  ws += BN * 16;
    float* lossa = ws;  ws += 16;
    float* predbp= ws;  ws += 16;
    // f16 workspace region
    _Float16* hp = (_Float16*)ws;
    _Float16* feat16 = hp; hp += BN * 64;
    _Float16* tA16   = hp; hp += BN * HDIM;
    _Float16* tB16   = hp; hp += BN * HDIM;
    _Float16* xin16  = hp; hp += BN * HDIM;
    _Float16* h16    = hp; hp += BN * HDIM;
    _Float16* msgp16 = hp; hp += BN * HDIM;
    _Float16* msgv16 = hp; hp += BN * HDIM;
    _Float16* t2h    = hp; hp += (size_t)ROWS_MSG * HDIM;
    _Float16* p_in0  = hp; hp += 2 * 6 * 512;
    _Float16* p_in1  = hp; hp += 3 * 6 * 512;
    _Float16* p_in2  = hp; hp += 3 * 6 * 512;
    _Float16* p_in3  = hp; hp += 3 * 6 * 512;
    _Float16* p_msg0 = hp; hp += 6 * 6 * 512;
    _Float16* p_mw0  = hp; hp += 3 * 6 * 512;
    _Float16* p_mw1  = hp; hp += 3 * 6 * 512;
    _Float16* p_mw2  = hp; hp += 3 * 6 * 512;
    _Float16* p_wih  = hp; hp += 6 * 24 * 512;
    _Float16* p_whh  = hp; hp += 3 * 24 * 512;
    _Float16* p_pw   = hp; hp += 3 * 1 * 512;

    auto pack = [&](const float* W, int ldw, int kreal, int nreal, int Kch, int Nt,
                    _Float16* outp) {
        int total = Kch * Nt * 512;
        pack_w_kernel<<<(total + 255) / 256, 256, 0, stream>>>(W, ldw, kreal, nreal,
                                                               Kch, Nt, outp);
    };
    pack(in0_W,             96, 48,  96,  2, 6,  p_in0);
    pack(in_Ws + 0 * 9216,  96, 96,  96,  3, 6,  p_in1);
    pack(in_Ws + 1 * 9216,  96, 96,  96,  3, 6,  p_in2);
    pack(in_Ws + 2 * 9216,  96, 96,  96,  3, 6,  p_in3);
    pack(msg0_W,            96, 192, 96,  6, 6,  p_msg0);
    pack(msg_Ws + 0 * 9216, 96, 96,  96,  3, 6,  p_mw0);
    pack(msg_Ws + 1 * 9216, 96, 96,  96,  3, 6,  p_mw1);
    pack(msg_Ws + 2 * 9216, 96, 96,  96,  3, 6,  p_mw2);
    pack(W_ih,             384, 192, 384, 6, 24, p_wih);
    pack(W_hh,             384, 96,  384, 3, 24, p_whh);
    pack(pred_W,             8, 96,  8,   3, 1,  p_pw);
    pad_bias_kernel<<<1, 16, 0, stream>>>(pred_b, 8, predbp);

    encode_kernel<<<(BN * 64 + 255) / 256, 256, 0, stream>>>(x, digit_emb, row_emb,
                                                             col_emb, feat16);
    init_hs_kernel<<<(BN * HDIM + 255) / 256, 256, 0, stream>>>(h16, sbuf, BN * HDIM);

    auto gemm = [&](const _Float16* X, int ldx, const _Float16* Wp, const float* bias,
                    float bscale, float* Yf, _Float16* Yh, int ldy,
                    int Mtiles, int Ntiles, int Kch, int relu) {
        int waves = Mtiles * Ntiles;
        gemm_kernel<<<(waves + 7) / 8, 256, 0, stream>>>(
            X, ldx, Wp, bias, bscale, Yf, Yh, ldy, Mtiles, Ntiles, Kch, relu);
    };

    // input encoder MLP: feat(48 pad 64)->96 ->96 ->96 ->96
    gemm(feat16, 64, p_in0, in0_b,       1.f, nullptr, tA16,  96, 64, 6, 2, 1);
    gemm(tA16,   96, p_in1, in_bs + 0,   1.f, nullptr, tB16,  96, 64, 6, 3, 1);
    gemm(tB16,   96, p_in2, in_bs + 96,  1.f, nullptr, tA16,  96, 64, 6, 3, 1);
    gemm(tA16,   96, p_in3, in_bs + 192, 1.f, nullptr, xin16, 96, 64, 6, 3, 0);

    for (int step = 0; step < STEPS; ++step) {
        msg_mlp_kernel<<<ROWS_MSG / 16 / 8, 256, 0, stream>>>(
            h16, edges, p_msg0, msg0_b, p_mw0, p_mw1, msg_bs, t2h);
        degsum_kernel<<<(BN * HDIM + 255) / 256, 256, 0, stream>>>(t2h, msgp16);
        // msg = (sum_d t2) @ Ws[2] + DEG * bs[2]   (last layer is linear)
        gemm(msgp16, 96, p_mw2, msg_bs + 192, (float)DEG, nullptr, msgv16, 96,
             64, 6, 3, 0);
        lstm_gates_kernel<<<(64 * 24 + 7) / 8, 256, 0, stream>>>(
            msgv16, xin16, h16, p_wih, p_whh, b_ih, b_hh, gates);
        lstm_update_kernel<<<(BN * HDIM + 255) / 256, 256, 0, stream>>>(gates, sbuf, h16);
        gemm(h16, 96, p_pw, predbp, 1.f, logit, nullptr, 16, 64, 1, 3, 0);
        stats_kernel<<<1, 1024, 0, stream>>>(logit, target, out, lossa, step);
    }
}
